// TransformerXL_3917010174239
// MI455X (gfx1250) — compile-verified, matrix operand order checked by reading, not run on hardware
//
// Transformer-XL forward for MI455X (gfx1250), bf16 WMMA everywhere.
//
// Roofline: ~0.7 TFLOP of GEMMs vs ~1 GB of mandatory HBM traffic
// (23.3 TB/s -> ~43us memory floor; bf16 WMMA peak -> ~70us compute floor).
// => All matmuls go through v_wmma_f32_16x16x32_bf16 (fp32 accum).
// A-tiles are staged with GLOBAL_LOAD_ASYNC_TO_LDS_B128 (CDNA5 async path,
// ASYNCcnt-tracked), B-tiles with a VALU transpose-gather. All tile-load
// bounds guards are removed: every call site has M % 128 == 0 and all B reads
// provably stay inside their allocations (epilogue stores remain guarded).

#include <hip/hip_runtime.h>

typedef __attribute__((ext_vector_type(16))) __bf16 v16bf;
typedef __attribute__((ext_vector_type(8)))  float  v8f;
typedef __attribute__((ext_vector_type(4)))  int    v4i;

typedef __attribute__((address_space(1))) v4i* as1_v4i_p;  // global
typedef __attribute__((address_space(3))) v4i* as3_v4i_p;  // LDS

#define DEV __device__ __forceinline__

#if defined(__has_builtin)
#if __has_builtin(__builtin_amdgcn_global_load_async_to_lds_b128)
#define HAS_ASYNC_LDS 1
#endif
#endif
#ifndef HAS_ASYNC_LDS
#define HAS_ASYNC_LDS 0
#endif

DEV unsigned short f2bf(float x) {
  unsigned int u = __float_as_uint(x);
  unsigned int r = u + 0x7fffu + ((u >> 16) & 1u);
  return (unsigned short)(r >> 16);
}
DEV float bf2f(unsigned short s) { return __uint_as_float(((unsigned int)s) << 16); }

namespace {
constexpr int NV = 32000;   // vocab
constexpr int NL = 8;       // layers
constexpr int NH = 16;      // heads
constexpr int ED = 1024;    // model dim
constexpr int ID = 64;      // head dim
constexpr int FD = 4096;    // ffn dim
constexpr int SQ = 512;     // seq len
constexpr int PM = 512;     // memory len
constexpr int NB = 4;       // batch
constexpr int HID = NH * ID;          // 1024
constexpr int KL = PM + SQ;           // 1024 keys
constexpr long long MROWS = (long long)SQ * NB;  // 2048 query rows
constexpr long long KROWS = (long long)KL * NB;  // 4096 key rows
constexpr int LSTR = 20;    // LDS tile row stride in dwords (80B: 16B-aligned, conflict-free)
}

// ---------------------------------------------------------------- GEMM params
struct GemmP {
  const unsigned short* A; long long lda;  long long a_bs;      // A[m*lda + k] (+ z*a_bs)
  const unsigned short* B; long long bsk;  long long bsn;       // B[k*bsk + n*bsn]
  long long b_bs; int b_mod;                                    // (+ (b_mod? z%b_mod : z)*b_bs)
  float* C; unsigned short* Cbf; long long ldc; long long c_bs; // C[m*ldc + n] (+ z*c_bs)
  const float* bias;
  int M, N, K;
  int flags;  // 1 = add existing C, 2 = relu, 4 = bf16 output (Cbf)
};

// Block tile 128x128, 8 waves (wave32) each computing 64x32, K step 32.
// Requirements guaranteed by all call sites in kernel_launch:
//   M % 128 == 0; K % 32 == 0; lda, a_bs, k0 all even (4B align) and
//   (m*lda + k0) % 8 == 0 (16B align for async b128); all B reads in-bounds.
__global__ __launch_bounds__(256) void txl_wmma_gemm(GemmP p) {
  __shared__ unsigned int Al[128 * LSTR];  // A tile [128 rows][32 k] bf16
  __shared__ unsigned int Bl[128 * LSTR];  // B tile transposed [128 n][32 k] bf16

  const int tid  = threadIdx.x;
  const int wave = tid >> 5;
  const int lane = tid & 31;
  const int half = lane >> 4;
  const int l16  = lane & 15;
  const int bm = blockIdx.x * 128;
  const int bn = blockIdx.y * 128;
  const long long z = blockIdx.z;

  const unsigned short* A = p.A + z * p.a_bs;
  const long long zb = p.b_mod ? (z % (long long)p.b_mod) : z;
  const unsigned short* B = p.B + zb * p.b_bs;

  const int wm = (wave >> 2) * 64;   // wave row block within 128
  const int wn = (wave & 3) * 32;    // wave col block within 128

  v8f acc[4][2];
  const v8f vz = {0.f, 0.f, 0.f, 0.f, 0.f, 0.f, 0.f, 0.f};
#pragma unroll
  for (int mt = 0; mt < 4; ++mt)
#pragma unroll
    for (int nt = 0; nt < 2; ++nt) acc[mt][nt] = vz;

  for (int k0 = 0; k0 < p.K; k0 += 32) {
    // ---- stage A tile: 128 rows x 64B, via async DMA to LDS (b128 chunks)
#if HAS_ASYNC_LDS
#pragma unroll
    for (int i = 0; i < 2; ++i) {
      int c  = tid + i * 256;          // 512 chunks of 16B
      int r  = c >> 2;                 // 4 chunks per row
      int ch = c & 3;
      const unsigned short* g = A + (long long)(bm + r) * p.lda + (long long)k0 + ch * 8;
      __builtin_amdgcn_global_load_async_to_lds_b128(
          (as1_v4i_p)(void*)(unsigned short*)g,
          (as3_v4i_p)(void*)&Al[r * LSTR + ch * 4],
          0, 0);
    }
#else
#pragma unroll
    for (int i = 0; i < 8; ++i) {
      int e  = tid + i * 256;          // dword index 0..2047
      int r  = e >> 4;
      int cd = e & 15;
      long long off = (long long)(bm + r) * p.lda + (long long)k0 + cd * 2;
      Al[r * LSTR + cd] = *(const unsigned int*)(A + off);
    }
#endif
    // ---- stage B tile transposed: Bl[n][k], pack two k into one dword.
    // When bsk==1 the two 16-bit reads merge into a single b32 load.
#pragma unroll
    for (int i = 0; i < 8; ++i) {
      int e  = tid + i * 256;          // pair index 0..2047
      int nl = e >> 4;
      int kd = e & 15;
      long long o0 = (long long)(k0 + kd * 2) * p.bsk + (long long)(bn + nl) * p.bsn;
      unsigned int w = (unsigned int)B[o0] | ((unsigned int)B[o0 + p.bsk] << 16);
      Bl[nl * LSTR + kd] = w;
    }
#if HAS_ASYNC_LDS
    asm volatile("s_wait_asynccnt 0x0" ::: "memory");  // A-tile DMA complete
#endif
    __syncthreads();

    union Frag { v16bf v; unsigned int u[8]; };
    Frag af[4], bfr[2];
    // A frag per ISA 16-bit 16x32 layout: lane half selects k+8, vgpr v -> k pair
#pragma unroll
    for (int mt = 0; mt < 4; ++mt) {
      int row = wm + mt * 16 + l16;
#pragma unroll
      for (int vv = 0; vv < 8; ++vv) {
        int kd = (vv & 3) + half * 4 + (vv >> 2) * 8;
        af[mt].u[vv] = Al[row * LSTR + kd];
      }
    }
#pragma unroll
    for (int nt = 0; nt < 2; ++nt) {
      int row = wn + nt * 16 + l16;
#pragma unroll
      for (int vv = 0; vv < 8; ++vv) {
        int kd = (vv & 3) + half * 4 + (vv >> 2) * 8;
        bfr[nt].u[vv] = Bl[row * LSTR + kd];
      }
    }
#pragma unroll
    for (int mt = 0; mt < 4; ++mt)
#pragma unroll
      for (int nt = 0; nt < 2; ++nt)
        acc[mt][nt] = __builtin_amdgcn_wmma_f32_16x16x32_bf16(
            false, af[mt].v, false, bfr[nt].v, (short)0, acc[mt][nt], false, false);
    __syncthreads();
  }

  // ---- epilogue (C/D layout: vgpr r -> M = r + 8*half, N = lane&15)
#pragma unroll
  for (int mt = 0; mt < 4; ++mt) {
#pragma unroll
    for (int nt = 0; nt < 2; ++nt) {
      int gn = bn + wn + nt * 16 + l16;
      if (gn >= p.N) continue;
#pragma unroll
      for (int r = 0; r < 8; ++r) {
        int gm = bm + wm + mt * 16 + half * 8 + r;
        if (gm >= p.M) continue;
        float val = acc[mt][nt][r];
        if (p.bias) val += p.bias[gn];
        long long co = (long long)gm * p.ldc + gn;
        if (p.flags & 1) val += (p.C + z * p.c_bs)[co];
        if (p.flags & 2) val = val > 0.f ? val : 0.f;
        if (p.flags & 4) (p.Cbf + z * p.c_bs)[co] = f2bf(val);
        else             (p.C + z * p.c_bs)[co] = val;
      }
    }
  }
}

// ------------------------------------------------------------- block reduces
DEV float block_sum(float v, float* red) {
  int t = threadIdx.x;
  red[t] = v; __syncthreads();
  for (int o = 128; o > 0; o >>= 1) { if (t < o) red[t] += red[t + o]; __syncthreads(); }
  float r = red[0]; __syncthreads();
  return r;
}
DEV float block_max(float v, float* red) {
  int t = threadIdx.x;
  red[t] = v; __syncthreads();
  for (int o = 128; o > 0; o >>= 1) { if (t < o) red[t] = fmaxf(red[t], red[t + o]); __syncthreads(); }
  float r = red[0]; __syncthreads();
  return r;
}

// ------------------------------------------------------- elementwise kernels
__global__ __launch_bounds__(256) void txl_f2bf(const float* in, unsigned short* out, long long n) {
  long long i  = (long long)blockIdx.x * blockDim.x + threadIdx.x;
  long long st = (long long)gridDim.x * blockDim.x;
  for (; i < n; i += st) out[i] = f2bf(in[i]);
}

// x = emb[indices]; writes h (f32), h_bf, and new_memory[0]
__global__ __launch_bounds__(256) void txl_embed(const int* idx, const float* emb,
                                                 float* h, unsigned short* hbf, float* mem0) {
  long long n  = MROWS * ED;
  long long i  = (long long)blockIdx.x * blockDim.x + threadIdx.x;
  long long st = (long long)gridDim.x * blockDim.x;
  for (; i < n; i += st) {
    long long m = i / ED;
    int e = (int)(i % ED);
    float val = emb[(long long)idx[m] * ED + e];
    h[i] = val; hbf[i] = f2bf(val); mem0[i] = val;
  }
}

// sinusoidal relative positions, descending K-1..0, bf16 out (K, E)
__global__ __launch_bounds__(256) void txl_rpe(unsigned short* rpe) {
  long long n  = (long long)KL * ED;
  long long i  = (long long)blockIdx.x * blockDim.x + threadIdx.x;
  long long st = (long long)gridDim.x * blockDim.x;
  for (; i < n; i += st) {
    int k = (int)(i / ED), e = (int)(i % ED);
    float pos = (float)(KL - 1 - k);
    int j = (e < ED / 2) ? e : (e - ED / 2);
    float invf = __expf(-(2.0f * (float)j / (float)ED) * 9.210340371976184f); // ln(10000)
    float ang = pos * invf;
    rpe[i] = f2bf((e < ED / 2) ? sinf(ang) : cosf(ang));
  }
}

// cat = [memory_l ; h] flattened (K,B,E) -> bf16
__global__ __launch_bounds__(256) void txl_cat(const float* mem_l, const float* h,
                                               unsigned short* cat) {
  long long n    = KROWS * ED;
  long long splt = (long long)PM * NB * ED;
  long long i  = (long long)blockIdx.x * blockDim.x + threadIdx.x;
  long long st = (long long)gridDim.x * blockDim.x;
  for (; i < n; i += st)
    cat[i] = f2bf(i < splt ? mem_l[i] : h[i - splt]);
}

// qu = bf16(q + u[h,i]), qv = bf16(q + v[h,i]) over (S,B,H,I)
__global__ __launch_bounds__(256) void txl_quqv(const float* q, const float* u, const float* v,
                                                unsigned short* qu, unsigned short* qv) {
  long long n  = MROWS * HID;
  long long i  = (long long)blockIdx.x * blockDim.x + threadIdx.x;
  long long st = (long long)gridDim.x * blockDim.x;
  for (; i < n; i += st) {
    int hi = (int)(i % HID);
    float qq = q[i];
    qu[i] = f2bf(qq + u[hi]);
    qv[i] = f2bf(qq + v[hi]);
  }
}

// scores = where(mask, -1e9, (AC + rel_shift(BD)) / sqrt(I)); softmax -> bf16 p
// one block per (b,h,s) row of K=1024
__global__ __launch_bounds__(256) void txl_score_softmax(const float* AC, const float* BD,
                                                         unsigned short* P) {
  __shared__ float red[256];
  int rid = blockIdx.x;                 // b*H*S + h*S + s
  int s  = rid % SQ;
  int bh = rid / SQ;
  const float* ac  = AC + (long long)rid * KL;
  const float* bdb = BD + (long long)bh * SQ * KL;
  const float scale = 0.125f;           // 1/sqrt(64)
  float vals[4];
  float mx = -3.4e38f;
#pragma unroll
  for (int i = 0; i < 4; ++i) {
    int kk = threadIdx.x + i * 256;
    // rel_shift closed form: out[s,k] = y[f/(K+1), f%(K+1)], f = s*K + k + S
    int f  = s * KL + kk + SQ;
    int sy = f / (KL + 1);
    int jy = f % (KL + 1);
    float bd = (jy == 0) ? 0.f : bdb[(long long)sy * KL + (jy - 1)];
    float val = (kk > s + PM) ? -1e9f : (ac[kk] + bd) * scale;
    vals[i] = val;
    mx = fmaxf(mx, val);
  }
  mx = block_max(mx, red);
  float sum = 0.f;
#pragma unroll
  for (int i = 0; i < 4; ++i) { vals[i] = __expf(vals[i] - mx); sum += vals[i]; }
  sum = block_sum(sum, red);
  float inv = 1.f / sum;
#pragma unroll
  for (int i = 0; i < 4; ++i) {
    int kk = threadIdx.x + i * 256;
    P[(long long)rid * KL + kk] = f2bf(vals[i] * inv);
  }
}

// h = LN(h + add)*g + b; writes h (f32), h_bf, optional memory slot copy
__global__ __launch_bounds__(256) void txl_ln(float* h, const float* add, const float* g,
                                              const float* b, unsigned short* hbf,
                                              float* memout) {
  __shared__ float red[256];
  long long r = blockIdx.x;
  float x[4];
  float s = 0.f;
#pragma unroll
  for (int i = 0; i < 4; ++i) {
    int e = threadIdx.x + i * 256;
    float v = h[r * ED + e] + add[r * ED + e];
    x[i] = v; s += v;
  }
  float mean = block_sum(s, red) / (float)ED;
  float s2 = 0.f;
#pragma unroll
  for (int i = 0; i < 4; ++i) { float d = x[i] - mean; s2 += d * d; }
  float var = block_sum(s2, red) / (float)ED;
  float rstd = rsqrtf(var + 1e-5f);
#pragma unroll
  for (int i = 0; i < 4; ++i) {
    int e = threadIdx.x + i * 256;
    float y = (x[i] - mean) * rstd * g[e] + b[e];
    h[r * ED + e] = y;
    hbf[r * ED + e] = f2bf(y);
    if (memout) memout[r * ED + e] = y;
  }
}

// per-row: lse - logit[target]  (== -log_softmax picked at target)
__global__ __launch_bounds__(256) void txl_rowlse(const float* logits, const int* tgt,
                                                  float* rowloss) {
  __shared__ float red[256];
  long long m = blockIdx.x;
  const float* row = logits + m * NV;
  float mx = -3.4e38f;
  for (int nn = threadIdx.x; nn < NV; nn += 256) mx = fmaxf(mx, row[nn]);
  mx = block_max(mx, red);
  float s = 0.f;
  for (int nn = threadIdx.x; nn < NV; nn += 256) s += __expf(row[nn] - mx);
  s = block_sum(s, red);
  if (threadIdx.x == 0) rowloss[m] = mx + logf(s) - row[tgt[m]];
}

__global__ __launch_bounds__(256) void txl_lossred(const float* rowloss, float* out, int n) {
  __shared__ float red[256];
  float s = 0.f;
  for (int i = threadIdx.x; i < n; i += 256) s += rowloss[i];
  s = block_sum(s, red);
  if (threadIdx.x == 0) out[0] = s / (float)n;
}

// ---------------------------------------------------------------- host side
extern "C" void kernel_launch(void* const* d_in, const int* in_sizes, int n_in,
                              void* d_out_v, int out_size, void* d_ws, size_t ws_size,
                              hipStream_t stream) {
  const int*   indices = (const int*)  d_in[0];
  const int*   target  = (const int*)  d_in[1];
  const float* memory  = (const float*)d_in[2];
  const float* emb     = (const float*)d_in[3];
  const float* outbias = (const float*)d_in[4];
  const float* uu      = (const float*)d_in[5];
  const float* vv      = (const float*)d_in[6];
  const float* Wq      = (const float*)d_in[7];
  const float* Wk      = (const float*)d_in[8];
  const float* Wv      = (const float*)d_in[9];
  const float* Wr      = (const float*)d_in[10];
  const float* Wo      = (const float*)d_in[11];
  const float* ln1g    = (const float*)d_in[12];
  const float* ln1b    = (const float*)d_in[13];
  const float* W1      = (const float*)d_in[14];
  const float* b1      = (const float*)d_in[15];
  const float* W2      = (const float*)d_in[16];
  const float* b2      = (const float*)d_in[17];
  const float* ln2g    = (const float*)d_in[18];
  const float* ln2b    = (const float*)d_in[19];

  float* d_out    = (float*)d_out_v;
  float* d_logits = d_out + 1;
  float* d_newmem = d_out + 1 + (long long)SQ * NB * NV;

  // -------- workspace carve-out
  char* wsp = (char*)d_ws;
  size_t cur = 0;
  auto alloc = [&](size_t bytes) -> void* {
    void* p = wsp + cur;
    cur = (cur + bytes + 255) & ~(size_t)255;
    return p;
  };
  const size_t WPROJ = (size_t)NL * ED * HID;     // 8M elems per proj weight
  const size_t WFF   = (size_t)NL * ED * FD;      // 32M elems per ffn weight
  unsigned short* emb_bf = (unsigned short*)alloc((size_t)NV * ED * 2);
  unsigned short* Wq_bf  = (unsigned short*)alloc(WPROJ * 2);
  unsigned short* Wk_bf  = (unsigned short*)alloc(WPROJ * 2);
  unsigned short* Wv_bf  = (unsigned short*)alloc(WPROJ * 2);
  unsigned short* Wr_bf  = (unsigned short*)alloc(WPROJ * 2);
  unsigned short* Wo_bf  = (unsigned short*)alloc(WPROJ * 2);
  unsigned short* W1_bf  = (unsigned short*)alloc(WFF * 2);
  unsigned short* W2_bf  = (unsigned short*)alloc(WFF * 2);
  float*          hbuf   = (float*)alloc(MROWS * ED * 4);
  unsigned short* h_bf   = (unsigned short*)alloc(MROWS * ED * 2);
  float*          tmp    = (float*)alloc(MROWS * ED * 4);
  unsigned short* cat_bf = (unsigned short*)alloc(KROWS * ED * 2);
  unsigned short* rpe_bf = (unsigned short*)alloc((size_t)KL * ED * 2);
  float*          qbuf   = (float*)alloc(MROWS * HID * 4);
  unsigned short* qu_bf  = (unsigned short*)alloc(MROWS * HID * 2);
  unsigned short* qv_bf  = (unsigned short*)alloc(MROWS * HID * 2);
  unsigned short* k_bf   = (unsigned short*)alloc(KROWS * HID * 2);
  unsigned short* v_bf   = (unsigned short*)alloc(KROWS * HID * 2);
  unsigned short* rp_bf  = (unsigned short*)alloc((size_t)KL * HID * 2);
  float*          ACb    = (float*)alloc((size_t)NB * NH * SQ * KL * 4);
  float*          BDb    = (float*)alloc((size_t)NB * NH * SQ * KL * 4);
  unsigned short* p_bf   = (unsigned short*)alloc((size_t)NB * NH * SQ * KL * 2);
  float*          attnf  = (float*)alloc(MROWS * HID * 4);
  unsigned short* at_bf  = (unsigned short*)alloc(MROWS * HID * 2);
  unsigned short* f_bf   = (unsigned short*)alloc(MROWS * FD * 2);
  float*          rowls  = (float*)alloc(MROWS * 4);
  (void)ws_size; (void)n_in; (void)in_sizes; (void)out_size;

  auto conv = [&](const float* src, unsigned short* dst, long long n) {
    txl_f2bf<<<2048, 256, 0, stream>>>(src, dst, n);
  };
  auto gemm = [&](const unsigned short* A, long long lda, long long a_bs,
                  const unsigned short* B, long long bsk, long long bsn,
                  long long b_bs, int b_mod,
                  float* C, unsigned short* Cb, long long ldc, long long c_bs,
                  const float* bias, int M, int N, int K, int flags, int batch) {
    GemmP p;
    p.A = A; p.lda = lda; p.a_bs = a_bs;
    p.B = B; p.bsk = bsk; p.bsn = bsn; p.b_bs = b_bs; p.b_mod = b_mod;
    p.C = C; p.Cbf = Cb; p.ldc = ldc; p.c_bs = c_bs;
    p.bias = bias; p.M = M; p.N = N; p.K = K; p.flags = flags;
    dim3 g((M + 127) / 128, (N + 127) / 128, batch);
    txl_wmma_gemm<<<g, 256, 0, stream>>>(p);
  };

  // -------- one-time conversions
  conv(emb, emb_bf, (long long)NV * ED);
  conv(Wq, Wq_bf, WPROJ); conv(Wk, Wk_bf, WPROJ); conv(Wv, Wv_bf, WPROJ);
  conv(Wr, Wr_bf, WPROJ); conv(Wo, Wo_bf, WPROJ);
  conv(W1, W1_bf, WFF);   conv(W2, W2_bf, WFF);

  // -------- embeddings + positional encodings
  txl_embed<<<2048, 256, 0, stream>>>(indices, emb, hbuf, h_bf, d_newmem);
  txl_rpe<<<2048, 256, 0, stream>>>(rpe_bf);

  const int BH = NB * NH;  // 64 attention batches
  for (int l = 0; l < NL; ++l) {
    const long long wpo = (long long)l * ED * HID;
    const long long wfo = (long long)l * ED * FD;

    // cat = [memory_l ; h]  (K*B rows, E cols) bf16
    txl_cat<<<2048, 256, 0, stream>>>(memory + (long long)l * PM * NB * ED, hbuf, cat_bf);

    // q = h @ Wq  (2048 x 1024 x 1024) -> fp32, then +u/+v in bf16
    gemm(h_bf, ED, 0, Wq_bf + wpo, HID, 1, 0, 0, qbuf, nullptr, HID, 0,
         nullptr, (int)MROWS, HID, ED, 0, 1);
    txl_quqv<<<2048, 256, 0, stream>>>(qbuf, uu, vv, qu_bf, qv_bf);

    // k, v = cat @ Wk/Wv (4096 x 1024 x 1024) -> bf16 directly
    gemm(cat_bf, ED, 0, Wk_bf + wpo, HID, 1, 0, 0, nullptr, k_bf, HID, 0,
         nullptr, (int)KROWS, HID, ED, 4, 1);
    gemm(cat_bf, ED, 0, Wv_bf + wpo, HID, 1, 0, 0, nullptr, v_bf, HID, 0,
         nullptr, (int)KROWS, HID, ED, 4, 1);
    // rp = rpe @ Wr (1024 x 1024 x 1024) -> bf16
    gemm(rpe_bf, ED, 0, Wr_bf + wpo, HID, 1, 0, 0, nullptr, rp_bf, HID, 0,
         nullptr, KL, HID, ED, 4, 1);

    // AC[z=(b,h)] = (q+u)(S x I) @ k^T(I x K): B accessed transposed via strides
    gemm(qu_bf, (long long)NB * HID, ID,
         k_bf, 1, (long long)NB * HID, ID, 0,
         ACb, nullptr, KL, (long long)SQ * KL,
         nullptr, SQ, KL, ID, 0, BH);
    // BD[z] = (q+v)(S x I) @ rp^T(I x K); rp batch depends only on h -> b_mod=NH
    gemm(qv_bf, (long long)NB * HID, ID,
         rp_bf, 1, HID, ID, NH,
         BDb, nullptr, KL, (long long)SQ * KL,
         nullptr, SQ, KL, ID, 0, BH);

    // rel_shift + mask + scale + softmax -> p (bf16)
    txl_score_softmax<<<BH * SQ, 256, 0, stream>>>(ACb, BDb, p_bf);

    // attn[z] = p(S x K) @ v(K x I) -> (S,B,HI) fp32, then bf16
    gemm(p_bf, KL, (long long)SQ * KL,
         v_bf, (long long)NB * HID, 1, ID, 0,
         attnf, nullptr, (long long)NB * HID, ID,
         nullptr, SQ, ID, KL, 0, BH);
    conv(attnf, at_bf, MROWS * HID);

    // out-proj + residual LN1
    gemm(at_bf, HID, 0, Wo_bf + wpo, ED, 1, 0, 0, tmp, nullptr, ED, 0,
         nullptr, (int)MROWS, ED, HID, 0, 1);
    txl_ln<<<(int)MROWS, 256, 0, stream>>>(hbuf, tmp, ln1g + (long long)l * ED,
                                           ln1b + (long long)l * ED, h_bf, nullptr);

    // FFN: relu(h@W1 + b1) -> bf16; @W2 + b2 -> tmp; residual LN2 (+ new_memory)
    gemm(h_bf, ED, 0, W1_bf + wfo, FD, 1, 0, 0, nullptr, f_bf, FD, 0,
         b1 + (long long)l * FD, (int)MROWS, FD, ED, 2 | 4, 1);
    gemm(f_bf, FD, 0, W2_bf + wfo, ED, 1, 0, 0, tmp, nullptr, ED, 0,
         b2 + (long long)l * ED, (int)MROWS, ED, FD, 0, 1);
    txl_ln<<<(int)MROWS, 256, 0, stream>>>(hbuf, tmp, ln2g + (long long)l * ED,
                                           ln2b + (long long)l * ED, h_bf,
                                           d_newmem + (long long)(l + 1) * PM * NB * ED);
  }

  // logits = h @ emb^T + out_bias  (2048 x 32000 x 1024)
  gemm(h_bf, ED, 0, emb_bf, 1, ED, 0, 0, d_logits, nullptr, NV, 0,
       outbias, (int)MROWS, NV, ED, 0, 1);

  // loss = mean(lse - logit[target])
  txl_rowlse<<<(int)MROWS, 256, 0, stream>>>(d_logits, target, rowls);
  txl_lossred<<<1, 256, 0, stream>>>(rowls, d_out, (int)MROWS);
}
